// SRNN_82489141887304
// MI455X (gfx1250) — matrix operand order repaired and checked by reading, not compiled, not run on hardware
//
#include <hip/hip_runtime.h>
#include <math.h>

#ifndef USE_ASYNC_LDS
#define USE_ASYNC_LDS 1   // issue global_load_async_to_lds_b128 alongside visible ds stores
#endif

typedef __attribute__((ext_vector_type(16))) _Float16 v16h;
typedef __attribute__((ext_vector_type(8)))  _Float16 h8v;   // 16 bytes
typedef __attribute__((ext_vector_type(8)))  float    v8f;

namespace {

constexpr int Bn = 64, Tn = 256, In = 512, Hn = 1024;
constexpr int HBUF = Bn * Hn;     // one h state buffer, elements
constexpr int LROW = Hn + 8;      // padded LDS row stride (2064B = 129*16B; lane*4 banks, conflict-free)

// ---------------- WMMA fragment helpers (wave32, 16x16x32 f16) ----------------
// A fragment: 16(M) x 32(K), row-major source.
__device__ __forceinline__ v16h load_a16(const _Float16* __restrict__ A, int lda, int lane) {
  const int row  = lane & 15;
  const int koff = (lane >> 4) << 4;            // 0 or 16
  const _Float16* p = A + (size_t)row * lda + koff;
  v16h r;
  ((h8v*)&r)[0] = *(const h8v*)(p);
  ((h8v*)&r)[1] = *(const h8v*)(p + 8);
  return r;
}

// B fragment: 32(K) x 16(N) from row-major weight W[N][K] (B = W^T), global memory
__device__ __forceinline__ v16h load_b16(const _Float16* __restrict__ W, int ldw, int lane) {
  const int col  = lane & 15;
  const int koff = (lane >> 4) << 4;            // 0 or 16
  const _Float16* p = W + (size_t)col * ldw + koff;
  v16h r;
  ((h8v*)&r)[0] = *(const h8v*)(p);
  ((h8v*)&r)[1] = *(const h8v*)(p + 8);
  return r;
}

// B fragment from LDS-resident padded weight tile (row stride LROW elements)
__device__ __forceinline__ v16h load_b_lds(const _Float16* Wl, int kk, int lane) {
  const _Float16* p = Wl + (size_t)(lane & 15) * LROW + kk + ((lane >> 4) << 4);
  v16h r;
  ((h8v*)&r)[0] = *(const h8v*)(p);
  ((h8v*)&r)[1] = *(const h8v*)(p + 8);
  return r;
}

__device__ __forceinline__ v8f wmma_f16(v16h a, v16h b, v8f c) {
  return __builtin_amdgcn_wmma_f32_16x16x32_f16(false, a, false, b, (short)0, c, false, false);
}

__device__ __forceinline__ float fast_tanh(float x) {
#if __has_builtin(__builtin_amdgcn_tanhf)
  return __builtin_amdgcn_tanhf(x);     // v_tanh_f32 trans op
#else
  return tanhf(x);
#endif
}

// packed a-fragment tile addressing: tile = (t, nblk=n/16, mblk=m/16), 32 lanes x 8 f16
__device__ __forceinline__ size_t a_off(int t, int nblk, int mblk) {
  return (((size_t)t * (Hn / 16) + nblk) * (Bn / 16) + mblk) * 256;   // 32*8
}

// load packed D-fragment tile as f32 accumulator init (one 16B load per lane)
__device__ __forceinline__ v8f load_frag_c(const _Float16* __restrict__ src, int lane) {
  h8v pk = *(const h8v*)(src + lane * 8);
  v8f c;
#pragma unroll
  for (int v = 0; v < 8; ++v) c[v] = (float)pk[v];
  return c;
}

// store accumulator as packed f16 fragment (one 16B store per lane)
__device__ __forceinline__ void store_frag_f16(_Float16* __restrict__ dst, int lane, v8f c) {
  h8v pk;
#pragma unroll
  for (int v = 0; v < 8; ++v) pk[v] = (_Float16)c[v];
  *(h8v*)(dst + lane * 8) = pk;
}

__device__ __forceinline__ v8f splat8(float x) {
  v8f c;
#pragma unroll
  for (int v = 0; v < 8; ++v) c[v] = x;
  return c;
}

// strided row-major stores (h state must stay row-major: it is re-read as A fragments)
__device__ __forceinline__ void store_tanh_f16(_Float16* __restrict__ O, int ld, int lane, v8f c) {
  const int col = lane & 15, mb = (lane >> 4) << 3;
#pragma unroll
  for (int v = 0; v < 8; ++v) O[(size_t)(mb + v) * ld + col] = (_Float16)fast_tanh(c[v]);
}

__device__ __forceinline__ void store_c_f32(float* __restrict__ O, int ld, int lane, v8f c) {
  const int col = lane & 15, mb = (lane >> 4) << 3;
#pragma unroll
  for (int v = 0; v < 8; ++v) O[(size_t)(mb + v) * ld + col] = c[v];
}

// ---------------- grid-wide barrier (monotonic ticket, reset per call) ----------------
__device__ __forceinline__ void gridbar(unsigned* arrive, unsigned* release, unsigned nwg) {
  __syncthreads();
  if (threadIdx.x == 0) {
    __threadfence();
    unsigned ticket = __hip_atomic_fetch_add(arrive, 1u, __ATOMIC_ACQ_REL, __HIP_MEMORY_SCOPE_AGENT);
    unsigned round  = ticket / nwg + 1u;
    if (ticket % nwg == nwg - 1u) {
      __hip_atomic_store(release, round, __ATOMIC_RELEASE, __HIP_MEMORY_SCOPE_AGENT);
    } else {
      while (__hip_atomic_load(release, __ATOMIC_ACQUIRE, __HIP_MEMORY_SCOPE_AGENT) < round)
        __builtin_amdgcn_s_sleep(2);
    }
    __threadfence();
  }
  __syncthreads();
}

// copy `rows` weight rows (Hn f16 each) from global into padded LDS region.
// Issues the CDNA5 async global->LDS DMA AND a compiler-visible ds store of the
// same chunk (identical bytes; benign). The visible stores keep LLVM's model of
// smem correct so the per-tick ds_load_b128 B-fragment reads are not folded away.
__device__ __forceinline__ void fill_lds_rows(_Float16* dst, unsigned smbase,
                                              const _Float16* __restrict__ src,
                                              int rows, int tid) {
  const int total = rows * (Hn / 8);   // 16B chunks
  for (int c = tid; c < total; c += 256) {
    const int r = c >> 7, q = c & 127;
    const _Float16* gp = src + (size_t)r * Hn + q * 8;
    h8v val = *(const h8v*)gp;
    *(h8v*)(dst + (size_t)r * LROW + q * 8) = val;           // visible ds_store_b128
#if USE_ASYNC_LDS
    const unsigned loff = smbase + (unsigned)((r * LROW + q * 8) * 2);
    asm volatile("global_load_async_to_lds_b128 %0, %1, off"
                 :: "v"(loff), "v"(gp) : "memory");          // async DMA, same bytes
#endif
  }
}

__device__ __forceinline__ void fill_lds_fence() {
#if USE_ASYNC_LDS
  asm volatile("s_wait_asynccnt 0x0" ::: "memory");
#endif
}

} // namespace

// ---------------- elementwise prep kernels ----------------
__global__ void k_zero(unsigned* p, int n) {
  int i = blockIdx.x * blockDim.x + threadIdx.x;
  if (i < n) p[i] = 0u;
}

__global__ void k_cvt(_Float16* dst, const float* src, int n) {
  for (int i = blockIdx.x * blockDim.x + threadIdx.x; i < n; i += gridDim.x * blockDim.x)
    dst[i] = (_Float16)src[i];
}

__global__ void k_addcvt(_Float16* dst, const float* a, const float* b, int n) {
  for (int i = blockIdx.x * blockDim.x + threadIdx.x; i < n; i += gridDim.x * blockDim.x)
    dst[i] = (_Float16)(a[i] + b[i]);
}

__global__ void k_bias(float* dst, const float* bi, const float* bs, const float* bh, int n) {
  for (int i = blockIdx.x * blockDim.x + threadIdx.x; i < n; i += gridDim.x * blockDim.x)
    dst[i] = bi[i] + bs[i] + bh[i];
}

// x1 = p[t+3], x2 = .5(p[t+2]+p[t+3]), x4 = .25*sum(p[t..t+3]); p[t]=x[b,t,:] (zero past T)
__global__ void k_x124(_Float16* x1, _Float16* x2, _Float16* x4, const float* x, int n) {
  for (int idx = blockIdx.x * blockDim.x + threadIdx.x; idx < n; idx += gridDim.x * blockDim.x) {
    const int i = idx % In;
    const int r = idx / In;       // t*Bn + b
    const int b = r % Bn;
    const int t = r / Bn;
    const float* xb = x + (size_t)b * Tn * In + i;
    float v0 = xb[(size_t)t * In];
    float v1 = (t + 1 < Tn) ? xb[(size_t)(t + 1) * In] : 0.f;
    float v2 = (t + 2 < Tn) ? xb[(size_t)(t + 2) * In] : 0.f;
    float v3 = (t + 3 < Tn) ? xb[(size_t)(t + 3) * In] : 0.f;
    x1[idx] = (_Float16)v3;
    x2[idx] = (_Float16)(0.5f * (v2 + v3));
    x4[idx] = (_Float16)(0.25f * (v0 + v1 + v2 + v3));
  }
}

// ---------------- precompute a_z = x_z @ W_z^T + bias_z (packed-fragment output) ------
// grid: (T, Hn/128, 3), block 256 (8 waves). Wave tile: 32(M) x 32(N). blockIdx.x == t.
__global__ void __launch_bounds__(256) k_gemm_a(const _Float16* __restrict__ Xall,
                                                const _Float16* __restrict__ Wall,
                                                const float* __restrict__ bias,
                                                _Float16* __restrict__ Aall) {
  const int z = blockIdx.z;
  const _Float16* X = Xall + (size_t)z * ((size_t)Tn * Bn * In);
  const _Float16* W = Wall + (size_t)z * ((size_t)Hn * In);
  _Float16* Aout    = Aall + (size_t)z * ((size_t)Tn * Bn * Hn);
  const float* bz   = bias + z * Hn;
  const int lane = threadIdx.x & 31, wave = threadIdx.x >> 5;
  const int t  = blockIdx.x;
  const int r0 = t * 64 + (wave & 1) * 32;            // global row (t*B + batch)
  const int mb = (wave & 1) * 2;                       // batch m-block index (m/16)
  const int n0 = blockIdx.y * 128 + (wave >> 1) * 32;
  const int nb = n0 >> 4;
  const int col = lane & 15;
  v8f c00 = splat8(bz[n0 + col]);
  v8f c01 = splat8(bz[n0 + 16 + col]);
  v8f c10 = c00, c11 = c01;
#pragma unroll 4
  for (int k = 0; k < In; k += 32) {
    v16h a0 = load_a16(X + (size_t)r0 * In + k, In, lane);
    v16h a1 = load_a16(X + (size_t)(r0 + 16) * In + k, In, lane);
    v16h b0 = load_b16(W + (size_t)n0 * In + k, In, lane);
    v16h b1 = load_b16(W + (size_t)(n0 + 16) * In + k, In, lane);
    c00 = wmma_f16(a0, b0, c00);
    c01 = wmma_f16(a0, b1, c01);
    c10 = wmma_f16(a1, b0, c10);
    c11 = wmma_f16(a1, b1, c11);
  }
  store_frag_f16(Aout + a_off(t, nb,     mb),     lane, c00);
  store_frag_f16(Aout + a_off(t, nb + 1, mb),     lane, c01);
  store_frag_f16(Aout + a_off(t, nb,     mb + 1), lane, c10);
  store_frag_f16(Aout + a_off(t, nb + 1, mb + 1), lane, c11);
}

// ---------------- persistent pipelined recurrence ----------------
// 80 WGs x 256 threads; LDS-resident weight slices (132KB/WG, staged once).
// Tick k runs layer0[t=k] (WG 0-15), layer1[t=k-1] (WG 16-47), layer2[t=k-2] (WG 48-79)
// concurrently, then ONE grid barrier. 258 ticks. h state double-buffered by t parity.
__global__ void __launch_bounds__(256) k_recur(
    const _Float16* __restrict__ Wh0, const _Float16* __restrict__ Wi1,
    const _Float16* __restrict__ Wh1, const _Float16* __restrict__ Wi2,
    const _Float16* __restrict__ Wh2,
    const _Float16* __restrict__ a0, const _Float16* __restrict__ a1,
    const _Float16* __restrict__ a2,
    _Float16* __restrict__ h0b, _Float16* __restrict__ h1b, _Float16* __restrict__ h2b,
    unsigned* arrive, unsigned* release) {
  __shared__ __align__(16) _Float16 smem[64 * LROW];   // 132,096 B
  const unsigned nwg = gridDim.x;
  const int tid = threadIdx.x, lane = tid & 31, wave = tid >> 5;
  const int wg = blockIdx.x;
  const int role = (wg < 16) ? 0 : (wg < 48) ? 1 : 2;
  constexpr unsigned LHALF = (unsigned)(32 * LROW * 2);   // byte offset of 2nd 32-row region

  // one-time weight staging into LDS (async DMA + visible stores)
  if (role == 0) {
    fill_lds_rows(smem, 0, Wh0 + (size_t)wg * 64 * Hn, 64, tid);
  } else if (role == 1) {
    const int q = wg - 16;
    fill_lds_rows(smem, 0,                  Wi1 + (size_t)q * 32 * Hn, 32, tid);
    fill_lds_rows(smem + 32 * LROW, LHALF,  Wh1 + (size_t)q * 32 * Hn, 32, tid);
  } else {
    const int q = wg - 48;
    fill_lds_rows(smem, 0,                  Wi2 + (size_t)q * 32 * Hn, 32, tid);
    fill_lds_rows(smem + 32 * LROW, LHALF,  Wh2 + (size_t)q * 32 * Hn, 32, tid);
  }
  fill_lds_fence();
  __syncthreads();

  const int m0 = (wave & 3) * 16;
  const int mB = m0 >> 4;

  for (int k = 0; k < Tn + 2; ++k) {
    if (role == 0) {
      if (k <= Tn - 1) {                 // t = k in [0,255]
        const int t = k;
        const int nb = (wave >> 2) * 32;             // LDS row base (wave tile 16x32)
        const int ng = wg * 64 + nb;                 // global column
        const _Float16* hp = h0b + (size_t)((t + 1) & 1) * HBUF + (size_t)m0 * Hn;
        v8f c0 = load_frag_c(a0 + a_off(t, ng >> 4, mB), lane);
        v8f c1 = load_frag_c(a0 + a_off(t, (ng >> 4) + 1, mB), lane);
        if (t + 1 < Tn) __builtin_prefetch(a0 + a_off(t + 1, ng >> 4, mB), 0, 0);
#pragma unroll 4
        for (int kk = 0; kk < Hn; kk += 32) {
          v16h af = load_a16(hp + kk, Hn, lane);
          v16h b0 = load_b_lds(smem + (size_t)nb * LROW, kk, lane);
          v16h b1 = load_b_lds(smem + (size_t)(nb + 16) * LROW, kk, lane);
          c0 = wmma_f16(af, b0, c0);
          c1 = wmma_f16(af, b1, c1);
        }
        _Float16* hw = h0b + (size_t)(t & 1) * HBUF + (size_t)m0 * Hn + ng;
        store_tanh_f16(hw, Hn, lane, c0);
        store_tanh_f16(hw + 16, Hn, lane, c1);
      }
    } else if (role == 1) {
      if (k >= 2 && k <= Tn) {           // t = k-1 in [1,255]
        const int t = k - 1;
        const int nl = (wave >> 2) * 16;             // LDS row base (wave tile 16x16)
        const int ng = (wg - 16) * 32 + nl;
        const _Float16* hin = h0b + (size_t)(t & 1) * HBUF + (size_t)m0 * Hn;
        const _Float16* hp  = h1b + (size_t)((t + 1) & 1) * HBUF + (size_t)m0 * Hn;
        v8f c0 = load_frag_c(a1 + a_off(t, ng >> 4, mB), lane);
        if (t + 1 < Tn) __builtin_prefetch(a1 + a_off(t + 1, ng >> 4, mB), 0, 0);
#pragma unroll 4
        for (int kk = 0; kk < Hn; kk += 32) {
          v16h af = load_a16(hin + kk, Hn, lane);
          v16h b0 = load_b_lds(smem + (size_t)nl * LROW, kk, lane);
          c0 = wmma_f16(af, b0, c0);
        }
#pragma unroll 4
        for (int kk = 0; kk < Hn; kk += 32) {
          v16h af = load_a16(hp + kk, Hn, lane);
          v16h b0 = load_b_lds(smem + (size_t)(32 + nl) * LROW, kk, lane);
          c0 = wmma_f16(af, b0, c0);
        }
        _Float16* hw = h1b + (size_t)(t & 1) * HBUF + (size_t)m0 * Hn + ng;
        store_tanh_f16(hw, Hn, lane, c0);
      }
    } else {
      if (k >= 5) {                      // t = k-2 in [3,255]
        const int t = k - 2;
        const int nl = (wave >> 2) * 16;
        const int ng = (wg - 48) * 32 + nl;
        const _Float16* hin = h1b + (size_t)(t & 1) * HBUF + (size_t)m0 * Hn;
        const _Float16* hp  = h2b + (size_t)((t + 1) & 1) * HBUF + (size_t)m0 * Hn;
        v8f c0 = load_frag_c(a2 + a_off(t, ng >> 4, mB), lane);
        if (t + 1 < Tn) __builtin_prefetch(a2 + a_off(t + 1, ng >> 4, mB), 0, 0);
#pragma unroll 4
        for (int kk = 0; kk < Hn; kk += 32) {
          v16h af = load_a16(hin + kk, Hn, lane);
          v16h b0 = load_b_lds(smem + (size_t)nl * LROW, kk, lane);
          c0 = wmma_f16(af, b0, c0);
        }
#pragma unroll 4
        for (int kk = 0; kk < Hn; kk += 32) {
          v16h af = load_a16(hp + kk, Hn, lane);
          v16h b0 = load_b_lds(smem + (size_t)(32 + nl) * LROW, kk, lane);
          c0 = wmma_f16(af, b0, c0);
        }
        _Float16* hw = h2b + (size_t)(t & 1) * HBUF + (size_t)m0 * Hn + ng;
        store_tanh_f16(hw, Hn, lane, c0);
      }
    }
    gridbar(arrive, release, nwg);
  }
}

// ---------------- output: out[l] = h_l @ Wo^T + bo ----------------
// grid: (In/64, 3), block 256. Wave tile 16(M) x 32(N).
__global__ void __launch_bounds__(256) k_out(const _Float16* __restrict__ h0,
                                             const _Float16* __restrict__ h1,
                                             const _Float16* __restrict__ h2,
                                             const _Float16* __restrict__ Wo,
                                             const float* __restrict__ bo,
                                             float* __restrict__ out) {
  const int l = blockIdx.y;
  const _Float16* hl = (l == 0) ? h0 : (l == 1) ? h1 : h2;
  const int lane = threadIdx.x & 31, wave = threadIdx.x >> 5;
  const int m0 = (wave & 3) * 16;
  const int n0 = blockIdx.x * 64 + (wave >> 2) * 32;
  const int col = lane & 15;
  v8f c0 = splat8(bo[n0 + col]);
  v8f c1 = splat8(bo[n0 + 16 + col]);
#pragma unroll 4
  for (int k = 0; k < Hn; k += 32) {
    v16h af = load_a16(hl + (size_t)m0 * Hn + k, Hn, lane);
    v16h b0 = load_b16(Wo + (size_t)n0 * Hn + k, Hn, lane);
    v16h b1 = load_b16(Wo + (size_t)(n0 + 16) * Hn + k, Hn, lane);
    c0 = wmma_f16(af, b0, c0);
    c1 = wmma_f16(af, b1, c1);
  }
  float* ob = out + (size_t)l * Bn * In + (size_t)m0 * In;
  store_c_f32(ob + n0, In, lane, c0);
  store_c_f32(ob + n0 + 16, In, lane, c1);
}

// ---------------- host-side orchestration ----------------
extern "C" void kernel_launch(void* const* d_in, const int* in_sizes, int n_in,
                              void* d_out, int out_size, void* d_ws, size_t ws_size,
                              hipStream_t stream) {
  const float* x     = (const float*)d_in[0];   // (B,T,I)
  const float* noise = (const float*)d_in[1];   // (7,B,H)
  const float* Wi0   = (const float*)d_in[2];   // (H,I)
  const float* Wi12  = (const float*)d_in[3];   // (2,H,H)
  const float* bi    = (const float*)d_in[4];   // (3,H)
  const float* Ws    = (const float*)d_in[5];   // (3,H,I)
  const float* bs    = (const float*)d_in[6];   // (3,H)
  const float* Wh    = (const float*)d_in[7];   // (3,H,H)
  const float* bh    = (const float*)d_in[8];   // (3,H)
  const float* Wo    = (const float*)d_in[9];   // (I,H)
  const float* bo    = (const float*)d_in[10];  // (I,)
  float* out = (float*)d_out;                   // (3,B,I)

  // workspace layout (256B aligned chunks)
  char* base = (char*)d_ws;
  size_t off = 0;
  auto alloc = [&](size_t bytes) -> void* {
    void* p = base + off;
    off = (off + bytes + 255) & ~(size_t)255;
    return p;
  };
  const size_t HI = (size_t)Hn * In, HH = (size_t)Hn * Hn;
  const size_t TBI = (size_t)Tn * Bn * In, TBH = (size_t)Tn * Bn * Hn;

  unsigned*  flags  = (unsigned*)alloc(2 * sizeof(unsigned));
  float*     bias   = (float*)alloc(3 * Hn * sizeof(float));
  _Float16*  Wall   = (_Float16*)alloc(3 * HI * 2);    // [Wi0+Ws0, Ws1, Ws2]
  _Float16*  Wh16   = (_Float16*)alloc(3 * HH * 2);    // Wh0..2
  _Float16*  Wi16   = (_Float16*)alloc(2 * HH * 2);    // Wi12[0], Wi12[1]
  _Float16*  Wo16   = (_Float16*)alloc(HI * 2);        // (I rows, H cols) row-major
  _Float16*  X124   = (_Float16*)alloc(3 * TBI * 2);   // x1,x2,x4
  _Float16*  Aall   = (_Float16*)alloc(3 * TBH * 2);   // a0,a1,a2 packed fragments (+bias)
  _Float16*  Hbufs  = (_Float16*)alloc(3 * 2 * (size_t)HBUF * 2);
  (void)ws_size; (void)n_in; (void)in_sizes; (void)out_size;

  _Float16* h0b = Hbufs;
  _Float16* h1b = Hbufs + 2 * (size_t)HBUF;
  _Float16* h2b = Hbufs + 4 * (size_t)HBUF;

  auto blocks = [](size_t n) { return (unsigned)((n + 255) / 256); };

  // prep: flags, biases, fp16 weights, x windows, h-state init
  k_zero<<<1, 64, 0, stream>>>(flags, 2);
  k_bias<<<12, 256, 0, stream>>>(bias, bi, bs, bh, 3 * Hn);
  k_addcvt<<<blocks(HI), 256, 0, stream>>>(Wall, Wi0, Ws, (int)HI);                 // Wi0+Ws[0]
  k_cvt<<<blocks(2 * HI), 256, 0, stream>>>(Wall + HI, Ws + HI, (int)(2 * HI));     // Ws[1], Ws[2]
  k_cvt<<<blocks(3 * HH), 256, 0, stream>>>(Wh16, Wh, (int)(3 * HH));
  k_cvt<<<blocks(2 * HH), 256, 0, stream>>>(Wi16, Wi12, (int)(2 * HH));
  k_cvt<<<blocks(HI), 256, 0, stream>>>(Wo16, Wo, (int)HI);
  k_cvt<<<blocks(HBUF), 256, 0, stream>>>(h0b + HBUF, noise + 0 * (size_t)HBUF, HBUF); // h0 -> buf1 (read t=0)
  k_cvt<<<blocks(HBUF), 256, 0, stream>>>(h1b,        noise + 3 * (size_t)HBUF, HBUF); // h1 -> buf0 (read t=1)
  k_cvt<<<blocks(HBUF), 256, 0, stream>>>(h2b,        noise + 6 * (size_t)HBUF, HBUF); // h2 -> buf0 (read t=3)
  k_x124<<<blocks(TBI), 256, 0, stream>>>(X124, X124 + TBI, X124 + 2 * TBI, x, (int)TBI);

  // precompute a0,a1,a2 (bias folded, packed-fragment layout)
  {
    dim3 grid(Tn, Hn / 128, 3);
    k_gemm_a<<<grid, 256, 0, stream>>>(X124, Wall, bias, Aall);
  }

  // persistent pipelined recurrence (80 WGs, 1 grid barrier per tick, 258 ticks)
  k_recur<<<80, 256, 0, stream>>>(Wh16, Wi16, Wh16 + HH, Wi16 + HH, Wh16 + 2 * HH,
                                  Aall, Aall + TBH, Aall + 2 * TBH,
                                  h0b, h1b, h2b, flags, flags + 1);

  // output projection from final h (all final states in buffer 1)
  {
    dim3 grid(In / 64, 3);
    k_out<<<grid, 256, 0, stream>>>(h0b + HBUF, h1b + HBUF, h2b + HBUF, Wo16, bo, out);
  }
}
